// ConcatCharLSTM_LSTM_CRF_65292092834277
// MI455X (gfx1250) — compile-verified, matrix operand order checked by reading, not run on hardware
//
#include <hip/hip_runtime.h>
#include <hip/hip_bf16.h>
#include <stdint.h>
#include <stddef.h>

typedef __bf16 bf16;
typedef __attribute__((ext_vector_type(16))) __bf16 v16bf;
typedef __attribute__((ext_vector_type(8)))  __bf16 v8bf;
typedef __attribute__((ext_vector_type(8)))  float  v8f;

#define T_WORDS 2048
#define C_CHARS 8192
#define CDIM    256
#define CHD     128
#define CG      (4*CHD)    // 512
#define WDIM    1024
#define DIN     1536
#define WHD     512
#define WG      (4*WHD)    // 2048
#define NTAG    6
#define START_TAG 4
#define STOP_TAG  5
#define NEGV    (-10000.0f)

// ---------------------------------------------------------------------------
// Small data-movement kernels
// ---------------------------------------------------------------------------
__global__ void k_gather_chars(const int* __restrict__ chars,
                               const float* __restrict__ char_embed,
                               bf16* __restrict__ Xc) {
  int idx = blockIdx.x * blockDim.x + threadIdx.x;
  if (idx >= C_CHARS * CDIM) return;
  int i = idx / CDIM, d = idx % CDIM;
  Xc[idx] = (bf16)char_embed[(size_t)chars[i] * CDIM + d];
}

__global__ void k_f2bf(const float* __restrict__ src, bf16* __restrict__ dst, int n) {
  int i = blockIdx.x * blockDim.x + threadIdx.x;
  if (i < n) dst[i] = (bf16)src[i];
}

// W[G,H] (row-major) -> WT[H,G]
__global__ void k_transpose(const float* __restrict__ W, float* __restrict__ WT,
                            int G, int H) {
  int i = blockIdx.x * blockDim.x + threadIdx.x;
  if (i >= G * H) return;
  int g = i / H, h = i % H;
  WT[(size_t)h * G + g] = W[i];
}

// embeds[t] = [chF[s], chB[s], chF[e], chB[e], word_embed[words[t]]]  (bf16)
__global__ void k_build_embeds(const int* __restrict__ words,
                               const int* __restrict__ ix_seq,
                               const float* __restrict__ word_embed,
                               const float* __restrict__ chF,
                               const float* __restrict__ chB,
                               bf16* __restrict__ Xw) {
  int idx = blockIdx.x * blockDim.x + threadIdx.x;
  if (idx >= T_WORDS * DIN) return;
  int t = idx / DIN, d = idx % DIN;
  float v;
  if (d < 512) {
    int s = ix_seq[t];
    int e = ix_seq[t + 1] - 1;
    if (d < 128)      v = chF[(size_t)s * CHD + d];
    else if (d < 256) v = chB[(size_t)s * CHD + (d - 128)];
    else if (d < 384) v = chF[(size_t)e * CHD + (d - 256)];
    else              v = chB[(size_t)e * CHD + (d - 384)];
  } else {
    v = word_embed[(size_t)words[t] * WDIM + (d - 512)];
  }
  Xw[idx] = (bf16)v;
}

// ---------------------------------------------------------------------------
// WMMA GEMM, register-blocked:  C[M,N] (f32) = A[M,K](bf16,rm) * B[N,K](bf16,rm)^T
// One wave computes a (16*RB_M) x (16*RB_N) patch of C: RB_M*RB_N accumulators,
// per k-slab loads RB_M A-fragments + RB_N B-fragments and issues RB_M*RB_N
// v_wmma_f32_16x16x32_bf16  (8 WMMAs per 12 b128 loads at 2x4).
// __launch_bounds__(256,1): real block is 8 waves; give RA the full VGPR
// budget so the 8 v8f accumulators live in place (no spill / v_mov churn).
// Fragment layouts per CDNA5 ISA 7.12.2.
// ---------------------------------------------------------------------------
#define RB_M 2
#define RB_N 4

__global__ void __launch_bounds__(256, 1)
k_wmma_gemm_abt(const bf16* __restrict__ A,
                const bf16* __restrict__ B,
                float* __restrict__ C,
                int M, int N, int K) {
  const int lane = threadIdx.x & 31;
  const int wave = threadIdx.x >> 5;
  const int wpb  = blockDim.x >> 5;
  const int tile = blockIdx.x * wpb + wave;
  const int tilesN = N / (16 * RB_N);
  const int tilesM = M / (16 * RB_M);
  if (tile >= tilesM * tilesN) return;
  const int tm = (tile / tilesN) * (16 * RB_M);
  const int tn = (tile % tilesN) * (16 * RB_N);

  const int r  = lane & 15;            // row within 16-tile (A: M, B: N)
  const int ko = (lane >> 4) << 3;     // K sub-offset: 0 or 8

  const bf16* ap[RB_M];
  const bf16* bp[RB_N];
#pragma unroll
  for (int m = 0; m < RB_M; ++m)
    ap[m] = A + (size_t)(tm + m * 16 + r) * K + ko;
#pragma unroll
  for (int n = 0; n < RB_N; ++n)
    bp[n] = B + (size_t)(tn + n * 16 + r) * K + ko;

  v8f acc[RB_M][RB_N] = {};
  for (int k0 = 0; k0 < K; k0 += 32) {
    v16bf a[RB_M], b[RB_N];
#pragma unroll
    for (int m = 0; m < RB_M; ++m) {
      v8bf lo = *(const v8bf*)(ap[m] + k0);
      v8bf hi = *(const v8bf*)(ap[m] + k0 + 16);
      a[m] = __builtin_shufflevector(lo, hi, 0,1,2,3,4,5,6,7,8,9,10,11,12,13,14,15);
    }
#pragma unroll
    for (int n = 0; n < RB_N; ++n) {
      v8bf lo = *(const v8bf*)(bp[n] + k0);
      v8bf hi = *(const v8bf*)(bp[n] + k0 + 16);
      b[n] = __builtin_shufflevector(lo, hi, 0,1,2,3,4,5,6,7,8,9,10,11,12,13,14,15);
    }
#pragma unroll
    for (int m = 0; m < RB_M; ++m)
#pragma unroll
      for (int n = 0; n < RB_N; ++n)
        acc[m][n] = __builtin_amdgcn_wmma_f32_16x16x32_bf16(
            /*neg_a=*/false, a[m], /*neg_b=*/false, b[n],
            /*c_mod=*/(short)0, acc[m][n],
            /*reuse_a=*/false, /*reuse_b=*/false);
  }
  // D layout: lane -> N = (lane&15); VGPR v -> M = v + 8*(lane>>4)
#pragma unroll
  for (int m = 0; m < RB_M; ++m) {
    const int rb = tm + m * 16 + ((lane >> 4) << 3);
#pragma unroll
    for (int n = 0; n < RB_N; ++n) {
      const int col = tn + n * 16 + (lane & 15);
#pragma unroll
      for (int v = 0; v < 8; ++v)
        C[(size_t)(rb + v) * N + col] = acc[m][n][v];
    }
  }
}

// ---------------------------------------------------------------------------
// Sequential LSTM recurrence.  blockIdx.x = direction (0 fwd, 1 bwd).
// xproj holds X @ Wih^T (no bias).  WhT is Whh transposed to [H,G] so lane g
// reads WhT[j*G + g] (coalesced, L2-resident).  h lives in LDS; c in regs of
// threads tid < H.  dyn-LDS = (H + G) floats.  Next timestep's xproj row is
// prefetched (global_prefetch_b8) to hide the only non-resident stream.
// ---------------------------------------------------------------------------
__global__ void __launch_bounds__(1024, 1)
k_lstm_rec(const float* __restrict__ xprojF, const float* __restrict__ xprojB,
           const float* __restrict__ WhTF,  const float* __restrict__ WhTB,
           const float* __restrict__ bihF,  const float* __restrict__ bhhF,
           const float* __restrict__ bihB,  const float* __restrict__ bhhB,
           const float* __restrict__ h0,    const float* __restrict__ c0,
           float* __restrict__ houtF,       float* __restrict__ houtB,
           int T, int H, int G) {
  extern __shared__ float sm[];
  float* hS = sm;        // [H]
  float* gS = sm + H;    // [G]
  const int dir = blockIdx.x;
  const float* xproj = dir ? xprojB : xprojF;
  const float* WhT   = dir ? WhTB   : WhTF;
  const float* bih   = dir ? bihB   : bihF;
  const float* bhh   = dir ? bhhB   : bhhF;
  float*       hout  = dir ? houtB  : houtF;

  const int tid = threadIdx.x;
  const int nth = blockDim.x;
  const int gpt = G / nth;

  float cR = 0.0f;
  if (tid < H) {
    hS[tid] = h0[dir * H + tid];
    cR      = c0[dir * H + tid];
  }
  __syncthreads();

  for (int s = 0; s < T; ++s) {
    const int t = dir ? (T - 1 - s) : s;
    if (s + 1 < T) {
      const int tn = dir ? (t - 1) : (t + 1);
      __builtin_prefetch(xproj + (size_t)tn * G + tid, 0, 1);
    }
    for (int q = 0; q < gpt; ++q) {
      const int g = tid + q * nth;
      float acc = xproj[(size_t)t * G + g] + bih[g] + bhh[g];
      const float* wp = WhT + g;
      for (int j = 0; j < H; ++j)
        acc = fmaf(hS[j], wp[(size_t)j * G], acc);
      gS[g] = acc;
    }
    __syncthreads();
    if (tid < H) {
      const float ig = gS[tid];
      const float fg = gS[H + tid];
      const float gg = gS[2 * H + tid];
      const float og = gS[3 * H + tid];
      const float si = 1.0f / (1.0f + expf(-ig));
      const float sf = 1.0f / (1.0f + expf(-fg));
      const float so = 1.0f / (1.0f + expf(-og));
      cR = sf * cR + si * tanhf(gg);
      const float hN = so * tanhf(cR);
      hS[tid] = hN;
      hout[(size_t)t * H + tid] = hN;
    }
    __syncthreads();
  }
}

// ---------------------------------------------------------------------------
// feats[t,tag] = concat(whF[t], whB[t]) . hid2tag_W[tag] + b[tag]
// ---------------------------------------------------------------------------
__global__ void k_feats(const float* __restrict__ whF, const float* __restrict__ whB,
                        const float* __restrict__ W,   const float* __restrict__ b,
                        float* __restrict__ feats) {
  int idx = blockIdx.x * blockDim.x + threadIdx.x;
  if (idx >= T_WORDS * NTAG) return;
  int t = idx / NTAG, tag = idx % NTAG;
  const float* w  = W + (size_t)tag * (2 * WHD);
  const float* hf = whF + (size_t)t * WHD;
  const float* hb = whB + (size_t)t * WHD;
  float acc = b[tag];
  for (int j = 0; j < WHD; ++j) acc = fmaf(hf[j], w[j], acc);
  for (int j = 0; j < WHD; ++j) acc = fmaf(hb[j], w[WHD + j], acc);
  feats[idx] = acc;
}

// ---------------------------------------------------------------------------
// 6-state Viterbi forward scan + backtrace. One small block.
// temp[j,i] = fv[i] + feat[j] + trans[j,i];  max/argmax over i.
// ---------------------------------------------------------------------------
__global__ void k_viterbi(const float* __restrict__ feats,
                          const float* __restrict__ trans,
                          int* __restrict__ bps,
                          int* __restrict__ out) {
  __shared__ float fv[NTAG], nfv[NTAG];
  __shared__ float tr[NTAG * NTAG];
  const int tid = threadIdx.x;
  if (tid < NTAG * NTAG) tr[tid] = trans[tid];
  if (tid < NTAG) fv[tid] = (tid == START_TAG) ? 0.0f : NEGV;
  __syncthreads();
  for (int t = 0; t < T_WORDS; ++t) {
    if (tid < NTAG) {
      float best = -3.0e38f; int bi = 0;
      for (int i = 0; i < NTAG; ++i) {
        const float v = fv[i] + tr[tid * NTAG + i];
        if (v > best) { best = v; bi = i; }
      }
      nfv[tid] = best + feats[t * NTAG + tid];
      bps[t * NTAG + tid] = bi;
    }
    __syncthreads();
    if (tid < NTAG) fv[tid] = nfv[tid];
    __syncthreads();
  }
  if (tid == 0) {
    float best = -3.0e38f; int bi = 0;
    for (int j = 0; j < NTAG; ++j) {
      const float v = fv[j] + tr[j * NTAG + STOP_TAG];
      if (v > best) { best = v; bi = j; }
    }
    int cur = bi;
    for (int t = T_WORDS - 1; t >= 0; --t) {
      out[t] = cur;
      cur = bps[t * NTAG + cur];
    }
  }
}

// ---------------------------------------------------------------------------
extern "C" void kernel_launch(void* const* d_in, const int* in_sizes, int n_in,
                              void* d_out, int out_size, void* d_ws, size_t ws_size,
                              hipStream_t stream) {
  const int*   words      = (const int*)  d_in[0];
  const int*   chars      = (const int*)  d_in[1];
  const int*   ix_seq     = (const int*)  d_in[2];
  const float* word_embed = (const float*)d_in[3];
  const float* char_embed = (const float*)d_in[4];
  const float* c_Wih_f = (const float*)d_in[5];
  const float* c_Whh_f = (const float*)d_in[6];
  const float* c_bih_f = (const float*)d_in[7];
  const float* c_bhh_f = (const float*)d_in[8];
  const float* c_Wih_b = (const float*)d_in[9];
  const float* c_Whh_b = (const float*)d_in[10];
  const float* c_bih_b = (const float*)d_in[11];
  const float* c_bhh_b = (const float*)d_in[12];
  const float* c_h0    = (const float*)d_in[13];
  const float* c_c0    = (const float*)d_in[14];
  const float* w_Wih_f = (const float*)d_in[15];
  const float* w_Whh_f = (const float*)d_in[16];
  const float* w_bih_f = (const float*)d_in[17];
  const float* w_bhh_f = (const float*)d_in[18];
  const float* w_Wih_b = (const float*)d_in[19];
  const float* w_Whh_b = (const float*)d_in[20];
  const float* w_bih_b = (const float*)d_in[21];
  const float* w_bhh_b = (const float*)d_in[22];
  const float* w_h0    = (const float*)d_in[23];
  const float* w_c0    = (const float*)d_in[24];
  const float* h2t_W   = (const float*)d_in[25];
  const float* h2t_b   = (const float*)d_in[26];
  const float* trans   = (const float*)d_in[27];

  // --- carve workspace ---
  char* wsp = (char*)d_ws;
  auto alloc = [&](size_t bytes) -> void* {
    void* p = (void*)wsp;
    wsp += (bytes + 255) & ~(size_t)255;
    return p;
  };
  bf16*  Xc      = (bf16*) alloc((size_t)C_CHARS * CDIM * sizeof(bf16));
  bf16*  cWf     = (bf16*) alloc((size_t)CG * CDIM * sizeof(bf16));
  bf16*  cWb     = (bf16*) alloc((size_t)CG * CDIM * sizeof(bf16));
  bf16*  Xw      = (bf16*) alloc((size_t)T_WORDS * DIN * sizeof(bf16));
  bf16*  wWf     = (bf16*) alloc((size_t)WG * DIN * sizeof(bf16));
  bf16*  wWb     = (bf16*) alloc((size_t)WG * DIN * sizeof(bf16));
  float* cWhTf   = (float*)alloc((size_t)CHD * CG * sizeof(float));
  float* cWhTb   = (float*)alloc((size_t)CHD * CG * sizeof(float));
  float* wWhTf   = (float*)alloc((size_t)WHD * WG * sizeof(float));
  float* wWhTb   = (float*)alloc((size_t)WHD * WG * sizeof(float));
  float* xp_cf   = (float*)alloc((size_t)C_CHARS * CG * sizeof(float));
  float* xp_cb   = (float*)alloc((size_t)C_CHARS * CG * sizeof(float));
  float* ch_f    = (float*)alloc((size_t)C_CHARS * CHD * sizeof(float));
  float* ch_b    = (float*)alloc((size_t)C_CHARS * CHD * sizeof(float));
  float* xp_wf   = (float*)alloc((size_t)T_WORDS * WG * sizeof(float));
  float* xp_wb   = (float*)alloc((size_t)T_WORDS * WG * sizeof(float));
  float* wh_f    = (float*)alloc((size_t)T_WORDS * WHD * sizeof(float));
  float* wh_b    = (float*)alloc((size_t)T_WORDS * WHD * sizeof(float));
  float* feats   = (float*)alloc((size_t)T_WORDS * NTAG * sizeof(float));
  int*   bps     = (int*)  alloc((size_t)T_WORDS * NTAG * sizeof(int));

  const int BT = 256;
  auto blocks = [](size_t n, int bt) { return (int)((n + bt - 1) / bt); };

  // 1) char embedding gather (f32 -> bf16)
  k_gather_chars<<<blocks((size_t)C_CHARS * CDIM, BT), BT, 0, stream>>>(chars, char_embed, Xc);

  // 2) weight conversions (bf16 for WMMA)
  k_f2bf<<<blocks((size_t)CG * CDIM, BT), BT, 0, stream>>>(c_Wih_f, cWf, CG * CDIM);
  k_f2bf<<<blocks((size_t)CG * CDIM, BT), BT, 0, stream>>>(c_Wih_b, cWb, CG * CDIM);
  k_f2bf<<<blocks((size_t)WG * DIN,  BT), BT, 0, stream>>>(w_Wih_f, wWf, WG * DIN);
  k_f2bf<<<blocks((size_t)WG * DIN,  BT), BT, 0, stream>>>(w_Wih_b, wWb, WG * DIN);

  // 3) Whh transposes for coalesced recurrence reads
  k_transpose<<<blocks((size_t)CG * CHD, BT), BT, 0, stream>>>(c_Whh_f, cWhTf, CG, CHD);
  k_transpose<<<blocks((size_t)CG * CHD, BT), BT, 0, stream>>>(c_Whh_b, cWhTb, CG, CHD);
  k_transpose<<<blocks((size_t)WG * WHD, BT), BT, 0, stream>>>(w_Whh_f, wWhTf, WG, WHD);
  k_transpose<<<blocks((size_t)WG * WHD, BT), BT, 0, stream>>>(w_Whh_b, wWhTb, WG, WHD);

  // 4) char input projections: [8192,256] x [512,256]^T  (WMMA, 2x4 blocked)
  {
    int tiles = (C_CHARS / (16 * RB_M)) * (CG / (16 * RB_N));   // 2048
    int grid  = (tiles + 7) / 8;                                // 8 waves / block
    k_wmma_gemm_abt<<<grid, 256, 0, stream>>>(Xc, cWf, xp_cf, C_CHARS, CG, CDIM);
    k_wmma_gemm_abt<<<grid, 256, 0, stream>>>(Xc, cWb, xp_cb, C_CHARS, CG, CDIM);
  }

  // 5) char recurrence, fwd+bwd concurrently (grid=2)
  k_lstm_rec<<<2, CG, (CHD + CG) * sizeof(float), stream>>>(
      xp_cf, xp_cb, cWhTf, cWhTb, c_bih_f, c_bhh_f, c_bih_b, c_bhh_b,
      c_h0, c_c0, ch_f, ch_b, C_CHARS, CHD, CG);

  // 6) assemble word-LSTM inputs (bf16)
  k_build_embeds<<<blocks((size_t)T_WORDS * DIN, BT), BT, 0, stream>>>(
      words, ix_seq, word_embed, ch_f, ch_b, Xw);

  // 7) word input projections: [2048,1536] x [2048,1536]^T  (WMMA, 2x4 blocked)
  {
    int tiles = (T_WORDS / (16 * RB_M)) * (WG / (16 * RB_N));   // 2048
    int grid  = (tiles + 7) / 8;
    k_wmma_gemm_abt<<<grid, 256, 0, stream>>>(Xw, wWf, xp_wf, T_WORDS, WG, DIN);
    k_wmma_gemm_abt<<<grid, 256, 0, stream>>>(Xw, wWb, xp_wb, T_WORDS, WG, DIN);
  }

  // 8) word recurrence, fwd+bwd concurrently (1024 thr, 2 gates/thread)
  k_lstm_rec<<<2, 1024, (WHD + WG) * sizeof(float), stream>>>(
      xp_wf, xp_wb, wWhTf, wWhTb, w_bih_f, w_bhh_f, w_bih_b, w_bhh_b,
      w_h0, w_c0, wh_f, wh_b, T_WORDS, WHD, WG);

  // 9) tag projection
  k_feats<<<blocks((size_t)T_WORDS * NTAG, BT), BT, 0, stream>>>(
      wh_f, wh_b, h2t_W, h2t_b, feats);

  // 10) Viterbi decode -> int32 tags
  k_viterbi<<<1, 64, 0, stream>>>(feats, trans, bps, (int*)d_out);
}
